// RGDTLayer_61564061221140
// MI455X (gfx1250) — compile-verified, matrix operand order checked by reading, not run on hardware
//
#include <hip/hip_runtime.h>
#include <hip/hip_bf16.h>

// ---------------------------------------------------------------------------
// Types
// ---------------------------------------------------------------------------
typedef _Float16 v8h  __attribute__((ext_vector_type(8)));
typedef _Float16 v16h __attribute__((ext_vector_type(16)));
typedef float    v8f  __attribute__((ext_vector_type(8)));
typedef float    f32x4 __attribute__((ext_vector_type(4)));

#define ALPHA 0.1f
#define SLOPE 0.2f

// ---------------------------------------------------------------------------
// LayerNorm rows: in f32 [M,256] -> out f16 [M,256]. One wave per row.
// ---------------------------------------------------------------------------
__global__ void ln_rows_kernel(const float* __restrict__ in,
                               const float* __restrict__ g,
                               const float* __restrict__ b,
                               _Float16* __restrict__ out, int M)
{
    const int wave = threadIdx.x >> 5;
    const int lane = threadIdx.x & 31;
    const int row  = blockIdx.x * (blockDim.x >> 5) + wave;
    if (row >= M) return;

    const f32x4* p = (const f32x4*)(in + (size_t)row * 256 + lane * 8);
    f32x4 x0 = p[0], x1 = p[1];
    float s = 0.f, q = 0.f;
    #pragma unroll
    for (int i = 0; i < 4; ++i) { s += x0[i] + x1[i]; q += x0[i]*x0[i] + x1[i]*x1[i]; }
    #pragma unroll
    for (int o = 16; o > 0; o >>= 1) { s += __shfl_xor(s, o, 32); q += __shfl_xor(q, o, 32); }
    const float mean = s * (1.0f / 256.0f);
    const float var  = q * (1.0f / 256.0f) - mean * mean;
    const float rstd = rsqrtf(var + 1e-5f);

    const f32x4* pg = (const f32x4*)(g + lane * 8);
    const f32x4* pb = (const f32x4*)(b + lane * 8);
    f32x4 g0 = pg[0], g1 = pg[1], b0 = pb[0], b1 = pb[1];
    v8h o8;
    #pragma unroll
    for (int i = 0; i < 4; ++i) {
        o8[i]     = (_Float16)((x0[i] - mean) * rstd * g0[i] + b0[i]);
        o8[i + 4] = (_Float16)((x1[i] - mean) * rstd * g1[i] + b1[i]);
    }
    *(v8h*)(out + (size_t)row * 256 + lane * 8) = o8;
}

// ---------------------------------------------------------------------------
// Transpose + convert weights: dst_f16[n*K + k] = src_f32[k*Ncols + n]
// ---------------------------------------------------------------------------
__global__ void tc_kernel(_Float16* __restrict__ dst, const float* __restrict__ src,
                          int K, int Ncols)
{
    const int idx = blockIdx.x * blockDim.x + threadIdx.x;
    const int total = K * Ncols;
    if (idx >= total) return;
    const int n = idx / K;
    const int k = idx - n * K;
    dst[idx] = (_Float16)src[(size_t)k * Ncols + n];
}

// ---------------------------------------------------------------------------
// WMMA GEMM: C[M,Nc] = A[M,K] x B[K,Nc], A f16 row-major, Bt f16 = B^T
// (Bt[n*K+k] = B[k][n]).
// Register blocking: each wave computes 2 M-tiles x 4 N-tiles (8 accumulators):
// each A fragment reused 4x, each B fragment 2x -> 1.5 b128 loads per WMMA.
// K is a template parameter so all fragment offsets are immediate-folded
// (B column stride 32*K bytes <= 98KB fits the 24-bit offset field), leaving
// only 3 live address pairs. __launch_bounds__(256,1) gives the allocator a
// full VGPR budget so the 8 accumulators + pipelined fragments never spill.
// Block = 8 waves (2 in M x 4 in N) = 64x256 macro-tile.
// Fragment-per-lane layout follows the documented 16-bit 16x32 scheme:
//   lane<16 : row/col = lane,    K chunks {k0..k0+7},{k0+16..k0+23}
//   lane>=16: row/col = lane-16, K chunks {k0+8..k0+15},{k0+24..k0+31}
// MODE 0: plain f32 store. MODE 1: relu(x+bias) -> f16. MODE 2: x+bias+res -> f32.
// ---------------------------------------------------------------------------
template<int MODE, int K>
__global__ __launch_bounds__(256, 1)
void gemm_wmma_kernel(const _Float16* __restrict__ A,
                      const _Float16* __restrict__ Bt,
                      int M, int Nc,
                      float* __restrict__ outF,
                      _Float16* __restrict__ outH,
                      const float* __restrict__ bias,
                      const float* __restrict__ res)
{
    constexpr int RB = K / 8;                       // v8h chunks per row
    const int lane  = threadIdx.x & 31;
    const int wave  = threadIdx.x >> 5;
    const int waveM = wave >> 2;                    // 0..1
    const int waveN = wave & 3;                     // 0..3
    const int l15   = lane & 15;
    const int hi    = lane >> 4;                    // 0 or 1

    const int mt0 = blockIdx.y * 4 + waveM * 2;     // first of 2 M-tiles
    const int nt0 = blockIdx.x * 16 + waveN * 4;    // first of 4 N-tiles
    if (nt0 * 16 >= Nc) return;                     // wave-uniform

    int r0 = mt0 * 16 + l15;       if (r0 >= M) r0 = M - 1;   // clamp loads
    int r1 = (mt0 + 1) * 16 + l15; if (r1 >= M) r1 = M - 1;   // (stores guarded)
    const v8h* Ar0 = (const v8h*)(A + (size_t)r0 * K);
    const v8h* Ar1 = (const v8h*)(A + (size_t)r1 * K);
    const v8h* Bb  = (const v8h*)(Bt + (size_t)(nt0 * 16 + l15) * K);

    v8f acc[2][4] = {};
    #pragma unroll 2
    for (int ks = 0; ks < K / 32; ++ks) {
        const int c0 = ks * 4 + hi;                 // 8-half chunk index in the row
        v16h af[2], bf[4];
        {
            v8h a0 = Ar0[c0], a1 = Ar0[c0 + 2];
            af[0] = __builtin_shufflevector(a0, a1, 0,1,2,3,4,5,6,7,8,9,10,11,12,13,14,15);
            v8h a2 = Ar1[c0], a3 = Ar1[c0 + 2];
            af[1] = __builtin_shufflevector(a2, a3, 0,1,2,3,4,5,6,7,8,9,10,11,12,13,14,15);
        }
        #pragma unroll
        for (int j = 0; j < 4; ++j) {
            v8h b0 = Bb[j * 16 * RB + c0], b1 = Bb[j * 16 * RB + c0 + 2];
            bf[j] = __builtin_shufflevector(b0, b1, 0,1,2,3,4,5,6,7,8,9,10,11,12,13,14,15);
        }
        #pragma unroll
        for (int i = 0; i < 2; ++i)
            #pragma unroll
            for (int j = 0; j < 4; ++j)
                acc[i][j] = __builtin_amdgcn_wmma_f32_16x16x32_f16(
                    false, af[i], false, bf[j], (short)0, acc[i][j], false, false);
    }

    #pragma unroll
    for (int i = 0; i < 2; ++i) {
        #pragma unroll
        for (int v = 0; v < 8; ++v) {
            const int row = (mt0 + i) * 16 + v + hi * 8;
            if (row >= M) continue;
            #pragma unroll
            for (int j = 0; j < 4; ++j) {
                const int col = (nt0 + j) * 16 + l15;
                const size_t o = (size_t)row * Nc + col;
                float val = acc[i][j][v];
                if (MODE == 0) {
                    outF[o] = val;
                } else if (MODE == 1) {
                    val += bias[col];
                    outH[o] = (_Float16)(val > 0.f ? val : 0.f);
                } else {
                    outF[o] = val + bias[col] + res[o];
                }
            }
        }
    }
}

// ---------------------------------------------------------------------------
// Misc elementwise helpers
// ---------------------------------------------------------------------------
__global__ void fill_u32_kernel(unsigned* __restrict__ p, unsigned v, int n)
{
    const int i = blockIdx.x * blockDim.x + threadIdx.x;
    if (i < n) p[i] = v;
}

__global__ void deg_kernel(const int* __restrict__ dst, float* __restrict__ deg, int E)
{
    const int i = blockIdx.x * blockDim.x + threadIdx.x;
    if (i < E) atomicAdd(&deg[dst[i]], 1.0f);
}

__global__ void log_kernel(float* __restrict__ deg, int n)
{
    const int i = blockIdx.x * blockDim.x + threadIdx.x;
    if (i < n) deg[i] = logf(deg[i]);
}

// ---------------------------------------------------------------------------
// Edge attention scores. One wave per edge; lane owns 8 contiguous channels,
// one head spans exactly 4 lanes.
// ---------------------------------------------------------------------------
__global__ void edge_score_kernel(const float* __restrict__ fhtf,
                                  const float* __restrict__ fr_rel,
                                  const float* __restrict__ attn,
                                  const float* __restrict__ logdeg,
                                  const int* __restrict__ src,
                                  const int* __restrict__ dst,
                                  const int* __restrict__ rid,
                                  float* __restrict__ escore, int E)
{
    const int gw   = (blockIdx.x * blockDim.x + threadIdx.x) >> 5;
    const int lane = threadIdx.x & 31;
    if (gw >= E) return;
    const int s = src[gw], d = dst[gw], r = rid[gw];

    const f32x4* ph = (const f32x4*)(fhtf + (size_t)s * 768 + lane * 8);
    const f32x4* pt = (const f32x4*)(fhtf + (size_t)d * 768 + 256 + lane * 8);
    const f32x4* pr = (const f32x4*)(fr_rel + (size_t)r * 256 + lane * 8);
    const f32x4* pa = (const f32x4*)(attn + lane * 8);
    f32x4 h0 = ph[0], h1 = ph[1];
    f32x4 t0 = pt[0], t1 = pt[1];
    f32x4 r0 = pr[0], r1 = pr[1];
    f32x4 w0 = pa[0], w1 = pa[1];

    float acc = 0.f;
    #pragma unroll
    for (int i = 0; i < 4; ++i) {
        float v = h0[i] * t0[i] * r0[i];
        v = v > 0.f ? v : SLOPE * v;
        acc += v * w0[i];
        float u = h1[i] * t1[i] * r1[i];
        u = u > 0.f ? u : SLOPE * u;
        acc += u * w1[i];
    }
    acc += __shfl_xor(acc, 1, 32);
    acc += __shfl_xor(acc, 2, 32);
    if ((lane & 3) == 0) {
        const float scale = logdeg[d] * (1.0f / 32.0f);   // log(deg)/D
        escore[(size_t)gw * 8 + (lane >> 2)] = acc * scale;
    }
}

// float -> monotone u32 key
__device__ __forceinline__ unsigned fkey(float f) {
    unsigned b = __float_as_uint(f);
    return (b & 0x80000000u) ? ~b : (b | 0x80000000u);
}

__global__ void segmax_kernel(const float* __restrict__ escore,
                              const int* __restrict__ dst,
                              unsigned* __restrict__ mkey, int E)
{
    const int i = blockIdx.x * blockDim.x + threadIdx.x;
    if (i >= E * 8) return;
    const int e = i >> 3, h = i & 7;
    atomicMax(&mkey[(size_t)dst[e] * 8 + h], fkey(escore[i]));
}

__global__ void decode_kernel(unsigned* __restrict__ mkey, int n)
{
    const int i = blockIdx.x * blockDim.x + threadIdx.x;
    if (i >= n) return;
    const unsigned k = mkey[i];
    mkey[i] = (k & 0x80000000u) ? (k ^ 0x80000000u) : ~k;   // back to float bits
}

__global__ void expsum_kernel(float* __restrict__ escore,
                              const float* __restrict__ m,
                              float* __restrict__ denom,
                              const int* __restrict__ dst, int E)
{
    const int i = blockIdx.x * blockDim.x + threadIdx.x;
    if (i >= E * 8) return;
    const int e = i >> 3, h = i & 7;
    const float ex = __expf(escore[i] - m[(size_t)dst[e] * 8 + h]);
    escore[i] = ex;
    atomicAdd(&denom[(size_t)dst[e] * 8 + h], ex);
}

__global__ void norm_kernel(float* __restrict__ escore,
                            const float* __restrict__ denom,
                            const int* __restrict__ dst, int E)
{
    const int i = blockIdx.x * blockDim.x + threadIdx.x;
    if (i >= E * 8) return;
    const int e = i >> 3, h = i & 7;
    // fold (1-ALPHA) into the normalized attention once
    escore[i] = escore[i] * (1.0f - ALPHA) / denom[(size_t)dst[e] * 8 + h];
}

// out[n*256+i] = ALPHA * feat0[n*ld + i]
__global__ void init_feat_kernel(float* __restrict__ out,
                                 const float* __restrict__ feat0, int ld, int total)
{
    const int idx = blockIdx.x * blockDim.x + threadIdx.x;
    if (idx >= total) return;
    const int n = idx >> 8, i = idx & 255;
    out[idx] = ALPHA * feat0[(size_t)n * ld + i];
}

// One wave per edge: featOut[dst] += a'(e,h) * featIn[src]
__global__ void scatter_kernel(const float* __restrict__ featIn, int ldin,
                               const float* __restrict__ a,
                               const int* __restrict__ src,
                               const int* __restrict__ dst,
                               float* __restrict__ featOut, int E)
{
    const int gw   = (blockIdx.x * blockDim.x + threadIdx.x) >> 5;
    const int lane = threadIdx.x & 31;
    if (gw >= E) return;
    const int s = src[gw], d = dst[gw];
    const float av = a[(size_t)gw * 8 + (lane >> 2)];
    const f32x4* pin = (const f32x4*)(featIn + (size_t)s * ldin + lane * 8);
    f32x4 v0 = pin[0], v1 = pin[1];
    float* po = featOut + (size_t)d * 256 + lane * 8;
    #pragma unroll
    for (int i = 0; i < 4; ++i) atomicAdd(po + i, v0[i] * av);
    #pragma unroll
    for (int i = 0; i < 4; ++i) atomicAdd(po + 4 + i, v1[i] * av);
}

__global__ void residual_kernel(const float* __restrict__ feat,
                                const float* __restrict__ ent,
                                float* __restrict__ rst, int total)
{
    const int i = blockIdx.x * blockDim.x + threadIdx.x;
    if (i < total) rst[i] = feat[i] + ent[i];
}

// ---------------------------------------------------------------------------
// Host side
// ---------------------------------------------------------------------------
extern "C" void kernel_launch(void* const* d_in, const int* in_sizes, int n_in,
                              void* d_out, int out_size, void* d_ws, size_t ws_size,
                              hipStream_t stream)
{
    const int N = 20000, F = 256, E = 320000, R = 100;

    const float* ent_feat = (const float*)d_in[0];
    const float* rel_feat = (const float*)d_in[1];
    const float* W_head   = (const float*)d_in[2];
    const float* W_tail   = (const float*)d_in[3];
    const float* W_ent    = (const float*)d_in[4];
    const float* W_rel    = (const float*)d_in[5];
    const float* attn     = (const float*)d_in[6];
    const float* ln_ent_g = (const float*)d_in[7];
    const float* ln_ent_b = (const float*)d_in[8];
    const float* ln_rel_g = (const float*)d_in[9];
    const float* ln_rel_b = (const float*)d_in[10];
    const float* ln_ff_g  = (const float*)d_in[11];
    const float* ln_ff_b  = (const float*)d_in[12];
    const float* W1       = (const float*)d_in[13];
    const float* b1       = (const float*)d_in[14];
    const float* W2       = (const float*)d_in[15];
    const float* b2       = (const float*)d_in[16];
    const int*   src      = (const int*)d_in[17];
    const int*   dst      = (const int*)d_in[18];
    const int*   rid      = (const int*)d_in[19];
    float* out = (float*)d_out;

    // bump allocator over d_ws
    char* base = (char*)d_ws;
    size_t off = 0;
    auto alloc = [&](size_t bytes) -> void* {
        off = (off + 255) & ~(size_t)255;
        void* p = base + off;
        off += bytes;
        return p;
    };

    _Float16* x16    = (_Float16*)alloc((size_t)N * F * 2);
    _Float16* rln16  = (_Float16*)alloc((size_t)112 * F * 2);
    _Float16* wcat_t = (_Float16*)alloc((size_t)768 * 256 * 2);
    _Float16* wrel_t = (_Float16*)alloc((size_t)256 * 256 * 2);
    _Float16* w1_t   = (_Float16*)alloc((size_t)1024 * 256 * 2);
    _Float16* w2_t   = (_Float16*)alloc((size_t)256 * 1024 * 2);
    float*    fhtf   = (float*)alloc((size_t)N * 768 * 4);   // [fh|ftl|fen]
    float*    fr_rel = (float*)alloc((size_t)112 * 256 * 4);
    float*    deg    = (float*)alloc((size_t)N * 4);
    float*    escore = (float*)alloc((size_t)E * 8 * 4);
    unsigned* mkey   = (unsigned*)alloc((size_t)N * 8 * 4);
    float*    denom  = (float*)alloc((size_t)N * 8 * 4);
    float*    featA  = (float*)alloc((size_t)N * F * 4);
    float*    featB  = (float*)alloc((size_t)N * F * 4);
    float*    rst    = (float*)alloc((size_t)N * F * 4);
    _Float16* y16    = (_Float16*)alloc((size_t)N * F * 2);
    _Float16* hid16  = (_Float16*)alloc((size_t)N * 1024 * 2);

    // 1) LayerNorms -> f16 GEMM inputs
    ln_rows_kernel<<<(N + 7) / 8, 256, 0, stream>>>(ent_feat, ln_ent_g, ln_ent_b, x16, N);
    ln_rows_kernel<<<(R + 7) / 8, 256, 0, stream>>>(rel_feat, ln_rel_g, ln_rel_b, rln16, R);

    // 2) Weight transpose/convert (B^T layout, f16)
    tc_kernel<<<(256 * 256 + 255) / 256, 256, 0, stream>>>(wcat_t,               W_head, 256, 256);
    tc_kernel<<<(256 * 256 + 255) / 256, 256, 0, stream>>>(wcat_t + 256 * 256,   W_tail, 256, 256);
    tc_kernel<<<(256 * 256 + 255) / 256, 256, 0, stream>>>(wcat_t + 512 * 256,   W_ent,  256, 256);
    tc_kernel<<<(256 * 256 + 255) / 256, 256, 0, stream>>>(wrel_t, W_rel, 256, 256);
    tc_kernel<<<(1024 * 256 + 255) / 256, 256, 0, stream>>>(w1_t, W1, 256, 1024);
    tc_kernel<<<(1024 * 256 + 255) / 256, 256, 0, stream>>>(w2_t, W2, 1024, 256);

    // 3) Fused projection GEMM: [N,256] x [256,768] -> fhtf ; rel GEMM
    //    grid: x = Nc/256, y = ceil(M/64)
    gemm_wmma_kernel<0, 256><<<dim3(3, (N + 63) / 64), 256, 0, stream>>>(
        x16, wcat_t, N, 768, fhtf, nullptr, nullptr, nullptr);
    gemm_wmma_kernel<0, 256><<<dim3(1, (R + 63) / 64), 256, 0, stream>>>(
        rln16, wrel_t, R, 256, fr_rel, nullptr, nullptr, nullptr);

    // 4) Degrees + log
    fill_u32_kernel<<<(N + 255) / 256, 256, 0, stream>>>((unsigned*)deg, 0u, N);
    fill_u32_kernel<<<(N * 8 + 255) / 256, 256, 0, stream>>>(mkey, 0u, N * 8);
    fill_u32_kernel<<<(N * 8 + 255) / 256, 256, 0, stream>>>((unsigned*)denom, 0u, N * 8);
    deg_kernel<<<(E + 255) / 256, 256, 0, stream>>>(dst, deg, E);
    log_kernel<<<(N + 255) / 256, 256, 0, stream>>>(deg, N);

    // 5) Edge scores + edge softmax (per dst node, per head)
    edge_score_kernel<<<E / 8, 256, 0, stream>>>(fhtf, fr_rel, attn, deg,
                                                 src, dst, rid, escore, E);
    segmax_kernel<<<(E * 8 + 255) / 256, 256, 0, stream>>>(escore, dst, mkey, E);
    decode_kernel<<<(N * 8 + 255) / 256, 256, 0, stream>>>(mkey, N * 8);
    expsum_kernel<<<(E * 8 + 255) / 256, 256, 0, stream>>>(escore, (const float*)mkey,
                                                           denom, dst, E);
    norm_kernel<<<(E * 8 + 255) / 256, 256, 0, stream>>>(escore, denom, dst, E);

    // 6) 5 PPR hops. feat0 = fen = fhtf column block 512..767 (ld 768).
    const float* fen = fhtf + 512;
    const float* fin = fen;
    int ldin = 768;
    float* outbuf = featA;
    for (int t = 0; t < 5; ++t) {
        outbuf = (t & 1) ? featB : featA;
        init_feat_kernel<<<(N * F + 255) / 256, 256, 0, stream>>>(outbuf, fen, 768, N * F);
        scatter_kernel<<<E / 8, 256, 0, stream>>>(fin, ldin, escore, src, dst, outbuf, E);
        fin = outbuf;
        ldin = 256;
    }
    // final feat is in featA (t = 4)

    // 7) Residual + FFN (pre-LN) + final residual
    residual_kernel<<<(N * F + 255) / 256, 256, 0, stream>>>(featA, ent_feat, rst, N * F);
    ln_rows_kernel<<<(N + 7) / 8, 256, 0, stream>>>(rst, ln_ff_g, ln_ff_b, y16, N);
    gemm_wmma_kernel<1, 256><<<dim3(4, (N + 63) / 64), 256, 0, stream>>>(
        y16, w1_t, N, 1024, nullptr, hid16, b1, nullptr);
    gemm_wmma_kernel<2, 1024><<<dim3(1, (N + 63) / 64), 256, 0, stream>>>(
        hid16, w2_t, N, 256, out, nullptr, b2, rst);
}
